// MemoryBankContrastLoss_19636590477419
// MI455X (gfx1250) — matrix operand order repaired and checked by reading, not compiled, not run on hardware
//
#include <hip/hip_runtime.h>
#include <hip/hip_bf16.h>

// ---------------- CDNA5 types ----------------
typedef _Float16 f16;
typedef __attribute__((ext_vector_type(8)))  _Float16 v8h;
typedef __attribute__((ext_vector_type(16))) _Float16 v16h;
typedef __attribute__((ext_vector_type(8)))  float    v8f;
typedef __attribute__((ext_vector_type(4)))  int      v4i;

#define AS1 __attribute__((address_space(1)))
#define AS3 __attribute__((address_space(3)))

// ---------------- problem constants ----------------
#define NCLS   20
#define MEM    512
#define MAXV   256
#define C_DIM  256
#define HW     16384          // 128*128
#define N_PIX  65536          // 4*128*128 pixels per tensor
#define ANCH_N 5120           // NCLS*MAXV
#define INV_TEMP 10.0f
#define AUX_IDX 254
#define NT     (ANCH_N / 16)  // 320 column tiles
#define B_PITCH 264           // 256 + 8 f16 pad -> 132 dwords/row, kills bank conflicts

// =====================================================================
// K1: L2-normalize proj tensors along C, emit f16 embeddings [2][N_PIX][C]
// =====================================================================
__global__ void k_normalize(const float* __restrict__ pm,
                            const float* __restrict__ pa,
                            f16* __restrict__ embd)
{
    const float* src = (blockIdx.y == 0) ? pm : pa;
    f16* dst = embd + (size_t)blockIdx.y * N_PIX * C_DIM;
    int p  = blockIdx.x * blockDim.x + threadIdx.x;
    int b  = p >> 14;
    int hw = p & (HW - 1);
    const float* base = src + (size_t)b * C_DIM * HW + hw;
    float ss = 0.0f;
    for (int c = 0; c < C_DIM; ++c) {
        float v = base[(size_t)c * HW];
        ss += v * v;
    }
    float inv = 1.0f / fmaxf(sqrtf(ss), 1e-12f);
    f16* drow = dst + (size_t)p * C_DIM;
    for (int c = 0; c < C_DIM; ++c)
        drow[c] = (f16)(base[(size_t)c * HW] * inv);
}

// =====================================================================
// K2: deterministic per-class selection (ordered wave32 ballot scan).
// =====================================================================
__global__ void k_select(const int* __restrict__ main_gt,
                         const int* __restrict__ aux_gt,
                         int* __restrict__ bidx, int* __restrict__ bcnt,
                         int* __restrict__ aidx, int* __restrict__ acnt)
{
    int c = blockIdx.x, mode = blockIdx.y, lane = threadIdx.x;
    int N, K; int* idxOut; int* cntOut;
    if (mode == 2) { N = 2 * N_PIX; K = MAXV; idxOut = aidx + c * MAXV; cntOut = acnt + c; }
    else           { N = N_PIX;     K = MEM;  idxOut = bidx + (mode * NCLS + c) * MEM;
                     cntOut = bcnt + mode * NCLS + c; }
    int cnt = 0;
    for (int base = 0; base < N && cnt < K; base += 32) {
        int i = base + lane;
        int lb = -1;
        if (i < N) {
            int j = i;
            const int* L;
            if      (mode == 0) L = main_gt;
            else if (mode == 1) L = aux_gt;
            else { if (j < N_PIX) L = main_gt; else { L = aux_gt; j -= N_PIX; } }
            lb = L[j];
            if (lb == AUX_IDX) lb = NCLS - 1;
        }
        bool m = (lb == c);
        unsigned long long bm = __ballot(m);
        int pre = __popcll(bm & ((1ull << lane) - 1ull));
        if (m && (cnt + pre) < K) idxOut[cnt + pre] = i;
        cnt += __popcll(bm);
    }
    if (lane == 0) *cntOut = (cnt < K) ? cnt : K;
}

// =====================================================================
// K3: EMA bank update into workspace copies (inputs never mutated) + norms
// =====================================================================
__global__ void k_update_bank(const float* __restrict__ main_bank,
                              const float* __restrict__ ema_bank,
                              const f16* __restrict__ embd,
                              const int* __restrict__ bidx,
                              const int* __restrict__ bcnt,
                              float* __restrict__ bank_new,
                              float* __restrict__ norms)
{
    int c = blockIdx.x, b = blockIdx.y, t = threadIdx.x;
    const float* oldb  = (b == 0) ? main_bank : ema_bank;
    const f16*   feats = embd + (size_t)b * N_PIX * C_DIM;
    float mom = (b == 0) ? 0.9f : 0.999f;
    int cnt = bcnt[b * NCLS + c];
    for (int j = t; j < MEM; j += blockDim.x) {
        const float* orow = oldb + ((size_t)c * MEM + j) * C_DIM;
        float* nrow = bank_new + (((size_t)b * NCLS + c) * MEM + j) * C_DIM;
        float ss = 0.0f;
        if (j < cnt) {
            int i = bidx[(b * NCLS + c) * MEM + j];
            const f16* frow = feats + (size_t)i * C_DIM;
            for (int k = 0; k < C_DIM; ++k) {
                float v = mom * orow[k] + (1.0f - mom) * (float)frow[k];
                ss += v * v;
            }
            float inv = 1.0f / fmaxf(sqrtf(ss), 1e-12f);
            for (int k = 0; k < C_DIM; ++k) {
                float v = mom * orow[k] + (1.0f - mom) * (float)frow[k];
                nrow[k] = v * inv;
            }
            norms[((size_t)b * NCLS + c) * MEM + j] = 1.0f;
        } else {
            for (int k = 0; k < C_DIM; ++k) { float v = orow[k]; ss += v * v; nrow[k] = v; }
            norms[((size_t)b * NCLS + c) * MEM + j] = sqrtf(ss);
        }
    }
}

// =====================================================================
// K4a: per-(bank,class) pick first MAXV slots with norm > 1e-6 (ordered)
// =====================================================================
__global__ void k_sample_select(const float* __restrict__ norms,
                                int* __restrict__ sidx, int* __restrict__ scnt)
{
    int c = blockIdx.x, b = blockIdx.y, lane = threadIdx.x;
    const float* nr = norms + ((size_t)b * NCLS + c) * MEM;
    int cnt = 0;
    for (int base = 0; base < MEM && cnt < MAXV; base += 32) {
        int j = base + lane;
        bool m = nr[j] > 1e-6f;
        unsigned long long bm = __ballot(m);
        int pre = __popcll(bm & ((1ull << lane) - 1ull));
        if (m && (cnt + pre) < MAXV) sidx[(b * NCLS + c) * MAXV + cnt + pre] = j;
        cnt += __popcll(bm);
    }
    if (lane == 0) scnt[b * NCLS + c] = (cnt < MAXV) ? cnt : MAXV;
}

// =====================================================================
// K4b: gather contrast rows -> dense f16 [2][ANCH_N][C] + column validity
// =====================================================================
__global__ void k_gather_contrast(const float* __restrict__ bank_new,
                                  const int* __restrict__ sidx,
                                  const int* __restrict__ scnt,
                                  f16* __restrict__ contr,
                                  float* __restrict__ cval)
{
    int row = blockIdx.x, b = blockIdx.y, ch = threadIdx.x;
    int c = row >> 8, v = row & 255;
    bool valid = v < scnt[b * NCLS + c];
    float val = 0.0f;
    if (valid) {
        int slot = sidx[(b * NCLS + c) * MAXV + v];
        val = bank_new[(((size_t)b * NCLS + c) * MEM + slot) * C_DIM + ch];
    }
    contr[((size_t)b * ANCH_N + row) * C_DIM + ch] = (f16)val;
    if (ch == 0) cval[(size_t)b * ANCH_N + row] = valid ? 1.0f : 0.0f;
}

// =====================================================================
// K4c: gather anchors -> dense f16 [ANCH_N][C] + anchor validity
// =====================================================================
__global__ void k_gather_anchors(const f16* __restrict__ embd,
                                 const int* __restrict__ aidx,
                                 const int* __restrict__ acnt,
                                 f16* __restrict__ anch,
                                 float* __restrict__ aval)
{
    int row = blockIdx.x, ch = threadIdx.x;
    int c = row >> 8, v = row & 255;
    bool valid = v < acnt[c];
    f16 val = (f16)0.0f;
    if (valid) {
        int i = aidx[c * MAXV + v];
        val = embd[(size_t)i * C_DIM + ch];
    }
    anch[(size_t)row * C_DIM + ch] = val;
    if (ch == 0) aval[row] = valid ? 1.0f : 0.0f;
}

// =====================================================================
// K5: fused WMMA GEMM + online InfoNCE.
// 4 waves / workgroup, 64 anchor rows per block. B tiles (16 x 256 f16)
// are staged into LDS with GLOBAL_LOAD_ASYNC_TO_LDS_B128 (double-buffered,
// ASYNCcnt-tracked) and shared by all 4 waves -> 4x less L2 traffic and
// async copy overlaps the WMMA pipe. Inner loop is fully branch-free:
// logits are bounded (unit vectors, 1/T = 10) so sum-exp uses a fixed
// shift exp(x-10); invalid columns encode as x = -1e38 (exp -> exact 0),
// making the cross-lane merge a pure add/max tree.
// =====================================================================
__global__ void __launch_bounds__(128)
k_infonce_wmma(const f16* __restrict__ anch,
               const f16* __restrict__ contr,
               const float* __restrict__ cval,
               const float* __restrict__ aval,
               float* __restrict__ contrib,
               float* __restrict__ vmarr)
{
    __shared__ f16   ldsB[2][16][B_PITCH];   // double-buffered B tile (+pad)
    __shared__ float s_cval[ANCH_N];         // column validity, staged once

    const int bb  = blockIdx.x;              // 0..79 : 64 anchor rows each
    const int b   = blockIdx.y;              // 0 = main bank, 1 = ema bank
    const int tid = threadIdx.x;
    const int lane = tid & 31;
    const int wid  = tid >> 5;               // wave 0..3 -> row sub-block
    const int l = lane & 15, h = lane >> 4;

    const f16*   cbase = contr + (size_t)b * ANCH_N * C_DIM;
    const float* cv    = cval  + (size_t)b * ANCH_N;

    // ---- stage column-validity into LDS (once) ----
    for (int i = tid; i < ANCH_N; i += 128) s_cval[i] = cv[i];

    // ---- load A fragments once: rows bb*64 + wid*16 + l ----
    const int rowbase = bb * 64 + wid * 16;
    const f16* arow = anch + (size_t)(rowbase + l) * C_DIM;
    v16h afrag[8];
#pragma unroll
    for (int kk = 0; kk < 8; ++kk) {
        v8h lo = *(const v8h*)(arow + kk * 32 + h * 8);
        v8h hi = *(const v8h*)(arow + kk * 32 + 16 + h * 8);
        afrag[kk] = __builtin_shufflevector(lo, hi, 0,1,2,3,4,5,6,7,8,9,10,11,12,13,14,15);
    }
    const int alab = rowbase >> 8;            // 16 | 256 => one class per wave

    // ---- async stage of a 16x256 f16 tile: 128 thr x 64B (4 x b128) ----
    auto stage = [&](int tile, int buf) {
        const int n   = tid >> 3;             // 0..15 tile row
        const int seg = tid & 7;              // 0..7  64B segment
        const f16* gsrc = cbase + ((size_t)(tile * 16 + n)) * C_DIM + seg * 32;
        f16* ldst = &ldsB[buf][n][seg * 32];
#pragma unroll
        for (int q = 0; q < 4; ++q)
            __builtin_amdgcn_global_load_async_to_lds_b128(
                (AS1 v4i*)(gsrc + q * 8), (AS3 v4i*)(ldst + q * 8), 0, 0);
    };

    float mmax[8], ssum[8], psum[8], npos[8];
#pragma unroll
    for (int r = 0; r < 8; ++r) { mmax[r] = -1e30f; ssum[r] = 0.f; psum[r] = 0.f; npos[r] = 0.f; }

    // prologue: tile 0 -> buf 0
    stage(0, 0);
    __builtin_amdgcn_s_wait_asynccnt(0);
    __syncthreads();

    for (int t = 0; t < NT; ++t) {
        const int buf = t & 1;
        if (t + 1 < NT) stage(t + 1, buf ^ 1);   // overlaps with compute below

        // ---- B fragments from LDS (ds_load_b128 pairs, padded pitch) ----
        v16h bfr[8];
#pragma unroll
        for (int kk = 0; kk < 8; ++kk) {
            v8h blo = *(const v8h*)(&ldsB[buf][l][kk * 32 + h * 8]);
            v8h bhi = *(const v8h*)(&ldsB[buf][l][kk * 32 + 16 + h * 8]);
            bfr[kk] = __builtin_shufflevector(blo, bhi, 0,1,2,3,4,5,6,7,8,9,10,11,12,13,14,15);
        }

        v8f acc = {};
#pragma unroll
        for (int kk = 0; kk < 8; ++kk)
            acc = __builtin_amdgcn_wmma_f32_16x16x32_f16(
                false, afrag[kk], false, bfr[kk], (short)0, acc, false, false);

        // ---- branch-free online update ----
        const int   ncol = t * 16 + l;
        const float cvn  = s_cval[ncol];
        const bool  valid = cvn > 0.0f;
        const bool  posC  = valid && ((ncol >> 8) == alab);
#pragma unroll
        for (int r = 0; r < 8; ++r) {
            float x = valid ? acc[r] * INV_TEMP : -1e38f;
            mmax[r]  = fmaxf(mmax[r], x);
            ssum[r] += __expf(x - 10.0f);        // fixed shift, exact 0 if invalid
            psum[r] += posC ? x : 0.0f;
            npos[r] += posC ? 1.0f : 0.0f;
        }

        if (t + 1 < NT) __builtin_amdgcn_s_wait_asynccnt(0);
        __syncthreads();
    }

    // ---- combine the 16 lanes sharing each row (pure add/max tree) ----
#pragma unroll
    for (int r = 0; r < 8; ++r) {
        float m = mmax[r], s = ssum[r], p = psum[r], n = npos[r];
#pragma unroll
        for (int w = 1; w < 16; w <<= 1) {
            m  = fmaxf(m, __shfl_xor(m, w, 16));
            s += __shfl_xor(s, w, 16);
            p += __shfl_xor(p, w, 16);
            n += __shfl_xor(n, w, 16);
        }
        if (l == 0) {
            int a = rowbase + (r + 8 * h);
            bool vm = (aval[a] > 0.0f) && (n > 0.0f);
            float se  = s * __expf(10.0f - m);   // = sum exp(x - max)
            float plp = (p - n * m - n * __logf(se + 1e-8f)) / (n + 1e-8f);
            contrib[(size_t)b * ANCH_N + a] = vm ? plp : 0.0f;
            vmarr  [(size_t)b * ANCH_N + a] = vm ? 1.0f : 0.0f;
        }
    }
}

// =====================================================================
// K6: final masked means -> scalar loss
// =====================================================================
__global__ void k_reduce(const float* __restrict__ contrib,
                         const float* __restrict__ vmarr,
                         float* __restrict__ out)
{
    __shared__ float s_sum[256], s_cnt[256];
    int t = threadIdx.x;
    float losses[2];
    for (int b = 0; b < 2; ++b) {
        float sum = 0.f, cnt = 0.f;
        for (int a = t; a < ANCH_N; a += 256) {
            sum += contrib[(size_t)b * ANCH_N + a];
            cnt += vmarr [(size_t)b * ANCH_N + a];
        }
        s_sum[t] = sum; s_cnt[t] = cnt;
        __syncthreads();
        for (int off = 128; off > 0; off >>= 1) {
            if (t < off) { s_sum[t] += s_sum[t + off]; s_cnt[t] += s_cnt[t + off]; }
            __syncthreads();
        }
        losses[b] = -s_sum[0] / fmaxf(s_cnt[0], 1.0f);
        __syncthreads();
    }
    if (t == 0) out[0] = 0.5f * losses[1] + 0.5f * losses[0]; // ema + main
}

// =====================================================================
// launcher
// =====================================================================
extern "C" void kernel_launch(void* const* d_in, const int* in_sizes, int n_in,
                              void* d_out, int out_size, void* d_ws, size_t ws_size,
                              hipStream_t stream)
{
    const float* main_proj = (const float*)d_in[0];
    const int*   main_gt   = (const int*)  d_in[1];
    const float* aux_proj  = (const float*)d_in[2];
    const int*   aux_gt    = (const int*)  d_in[3];
    const float* ema_bank  = (const float*)d_in[4];
    const float* main_bank = (const float*)d_in[5];
    float* out = (float*)d_out;

    char* ws = (char*)d_ws;
    size_t off = 0;
    auto take = [&](size_t bytes) -> char* {
        char* p = ws + off;
        off = (off + bytes + 255) & ~(size_t)255;
        return p;
    };
    f16*   embd     = (f16*)  take((size_t)2 * N_PIX * C_DIM * sizeof(f16));        // 64 MB
    float* bank_new = (float*)take((size_t)2 * NCLS * MEM * C_DIM * sizeof(float)); // 21 MB
    float* norms    = (float*)take((size_t)2 * NCLS * MEM * sizeof(float));
    int*   bidx     = (int*)  take((size_t)2 * NCLS * MEM * sizeof(int));
    int*   bcnt     = (int*)  take((size_t)2 * NCLS * sizeof(int));
    int*   aidx     = (int*)  take((size_t)NCLS * MAXV * sizeof(int));
    int*   acnt     = (int*)  take((size_t)NCLS * sizeof(int));
    int*   sidx     = (int*)  take((size_t)2 * NCLS * MAXV * sizeof(int));
    int*   scnt     = (int*)  take((size_t)2 * NCLS * sizeof(int));
    f16*   contr    = (f16*)  take((size_t)2 * ANCH_N * C_DIM * sizeof(f16));       // 5 MB
    float* cval     = (float*)take((size_t)2 * ANCH_N * sizeof(float));
    f16*   anch     = (f16*)  take((size_t)ANCH_N * C_DIM * sizeof(f16));           // 2.5 MB
    float* aval     = (float*)take((size_t)ANCH_N * sizeof(float));
    float* contrib  = (float*)take((size_t)2 * ANCH_N * sizeof(float));
    float* vmarr    = (float*)take((size_t)2 * ANCH_N * sizeof(float));
    (void)ws_size; (void)in_sizes; (void)n_in; (void)out_size;

    k_normalize      <<<dim3(N_PIX / 256, 2), 256, 0, stream>>>(main_proj, aux_proj, embd);
    k_select         <<<dim3(NCLS, 3),        32,  0, stream>>>(main_gt, aux_gt, bidx, bcnt, aidx, acnt);
    k_update_bank    <<<dim3(NCLS, 2),        256, 0, stream>>>(main_bank, ema_bank, embd, bidx, bcnt,
                                                                bank_new, norms);
    k_sample_select  <<<dim3(NCLS, 2),        32,  0, stream>>>(norms, sidx, scnt);
    k_gather_contrast<<<dim3(ANCH_N, 2),      256, 0, stream>>>(bank_new, sidx, scnt, contr, cval);
    k_gather_anchors <<<dim3(ANCH_N),         256, 0, stream>>>(embd, aidx, acnt, anch, aval);
    k_infonce_wmma   <<<dim3(ANCH_N / 64, 2), 128, 0, stream>>>(anch, contr, cval, aval, contrib, vmarr);
    k_reduce         <<<dim3(1),              256, 0, stream>>>(contrib, vmarr, out);
}
